// MhAttention_10582799417737
// MI455X (gfx1250) — matrix operand order repaired and checked by reading, not compiled, not run on hardware
//
#include <hip/hip_runtime.h>

typedef _Float16 half_t;
typedef __attribute__((ext_vector_type(16))) _Float16 v16h;
typedef __attribute__((ext_vector_type(8)))  _Float16 v8h;
typedef __attribute__((ext_vector_type(8)))  float    v8f;
typedef __attribute__((ext_vector_type(4)))  unsigned int v4u;
typedef __attribute__((ext_vector_type(8)))  int      v8i;
typedef __attribute__((ext_vector_type(4)))  int      v4i;

#define D_MODEL 1024
#define SEQ     2048
#define BATCH   2
#define NH      16
#define HD      64

#if defined(__has_builtin)
#  if __has_builtin(__builtin_amdgcn_tensor_load_to_lds)
#    define HAS_TDM 1
#  endif
#endif
#ifndef HAS_TDM
#  define HAS_TDM 0
#endif

// ---------- WMMA fragment helpers (16x16x32 f16, f32 accum) ----------
// A operand 16x32 (MxK): lane L holds row M=L%16; lanes 0-15: K={0..7,16..23},
// lanes 16-31: K={8..15,24..31}  (ISA 7.12.2 16-bit A-matrix table)
__device__ __forceinline__ v16h load_a_frag(const half_t* __restrict__ base,
                                            int ld, int row0, int k0) {
  const int lane = threadIdx.x & 31;
  const int l = lane & 15, hi = lane >> 4;
  const half_t* p = base + (size_t)(row0 + l) * ld + k0 + hi * 8;
  v8h x = *(const v8h*)p;
  v8h y = *(const v8h*)(p + 16);
  v16h r;
#pragma unroll
  for (int i = 0; i < 8; ++i) { r[i] = x[i]; r[i + 8] = y[i]; }
  return r;
}

// B operand 32x16 (KxN): lane L holds column N=L%16; lanes 0-15: K={0..7,16..23},
// lanes 16-31: K={8..15,24..31}. `base` must be N-major (base[n*ld + k]).
__device__ __forceinline__ v16h load_b_frag(const half_t* __restrict__ base,
                                            int ld, int col0, int k0) {
  const int lane = threadIdx.x & 31;
  const int l = lane & 15, hi = lane >> 4;
  const half_t* p = base + (size_t)(col0 + l) * ld + k0 + hi * 8;
  v8h x = *(const v8h*)p;
  v8h y = *(const v8h*)(p + 16);
  v16h r;
#pragma unroll
  for (int i = 0; i < 8; ++i) { r[i] = x[i]; r[i + 8] = y[i]; }
  return r;
}

__device__ __forceinline__ v8f wmma_f16(v16h a, v16h b, v8f c) {
  return __builtin_amdgcn_wmma_f32_16x16x32_f16(false, a, false, b,
                                                (short)0, c, false, false);
}

#if HAS_TDM
// Issue one TDM 2-D tile load: 32(k) x 128(m) fp16 slab of A into LDS.
// D# groups per ISA §8 (count=1 user desc, type=2, data_size=2B).
__device__ __forceinline__ void tdm_issue_a(const half_t* gtile, int K, int M,
                                            half_t* lds_ptr) {
  unsigned long long ga = (unsigned long long)(uintptr_t)gtile;
  unsigned int lds = (unsigned int)(uintptr_t)lds_ptr;
  v4u g0;
  g0[0] = 1u;                                   // count=1, user descriptor
  g0[1] = lds;                                  // lds_addr
  g0[2] = (unsigned int)(ga & 0xFFFFFFFFu);     // global_addr[31:0]
  g0[3] = (unsigned int)((ga >> 32) & 0x1FFFFFFu) | (2u << 30);  // type=2
  v8i g1;
  g1[0] = (int)(1u << 16);                      // data_size=1 (2 bytes)
  g1[1] = (int)(((unsigned)K & 0xFFFFu) << 16);             // tensor_dim0 lo16
  g1[2] = (int)(((unsigned)K >> 16) |
                (((unsigned)M & 0xFFFFu) << 16));           // dim0 hi | dim1 lo
  g1[3] = (int)(((unsigned)M >> 16) | (32u << 16));         // dim1 hi | tile_dim0
  g1[4] = (int)(128u);                          // tile_dim1=128, tile_dim2=0
  g1[5] = (int)K;                               // tensor_dim0_stride lo32
  g1[6] = 0;
  g1[7] = 0;
  v4i g2 = {0, 0, 0, 0};
  v4i g3 = {0, 0, 0, 0};
#if defined(__clang_major__) && (__clang_major__ >= 23)
  v8i g4 = {0, 0, 0, 0, 0, 0, 0, 0};
  __builtin_amdgcn_tensor_load_to_lds(g0, g1, g2, g3, g4, 0);
#else
  __builtin_amdgcn_tensor_load_to_lds(g0, g1, g2, g3, 0);
#endif
}
#endif

// ---------- weight transpose: out[n][k] = in[k][n], 1024x1024 fp16 ----------
__global__ __launch_bounds__(256) void transpose_w(const half_t* __restrict__ in,
                                                   half_t* __restrict__ out) {
  int idx = blockIdx.x * 256 + threadIdx.x;
  int k = idx >> 10, n = idx & 1023;
  out[((size_t)n << 10) | (size_t)k] = in[idx];
}

// ---------- V transpose: [b,s,h*64+d] -> Vt[b,h,d,s] ----------
__global__ __launch_bounds__(256) void v_transpose(const half_t* __restrict__ V,
                                                   half_t* __restrict__ Vt) {
  int idx = blockIdx.x * 256 + threadIdx.x;       // BATCH*SEQ*D_MODEL elems
  int c = idx & (D_MODEL - 1);
  int bs = idx >> 10;                             // b*SEQ + s
  int b = bs >> 11, s = bs & (SEQ - 1);
  int h = c >> 6, d = c & 63;
  Vt[((size_t)(b * NH + h) * HD + d) * SEQ + s] = V[idx];
}

// ---------- WMMA GEMM: C[M,N] = A[M,K] * Bt[N,K]^T ----------
// 256 threads (8 waves), block tile 128x128, wave tile 32x64.
// A k-slabs (128x32 fp16, 8 KB) are double-buffered in LDS and staged by the
// Tensor Data Mover: wave 0 waits TENSORcnt for the current slab, the block
// barriers, wave 0 issues the next slab's DMA, and all waves run WMMA on the
// current slab while the DMA is in flight. B streams from the 192 MB L2.
template <typename OUT>
__global__ __launch_bounds__(256) void gemm_wmma(const half_t* __restrict__ A,
                                                 const half_t* __restrict__ Bt,
                                                 OUT* __restrict__ C,
                                                 int M, int N, int K) {
  __shared__ __align__(16) half_t Ash[2][128 * 32];
  const int lane = threadIdx.x & 31;
  const int wave = threadIdx.x >> 5;
  const int mb = blockIdx.x * 128;
  const int mloc = (wave >> 1) * 32;              // wave's rows within LDS tile
  const int n0 = blockIdx.y * 128 + (wave & 1) * 64;
  v8f acc[2][4] = {};

  // prologue: stage the first slab
#if HAS_TDM
  if (wave == 0) tdm_issue_a(A + (size_t)mb * K, K, M, &Ash[0][0]);
#else
  {
    int t = threadIdx.x;
#pragma unroll
    for (int c = 0; c < 2; ++c) {
      int chunk = t * 2 + c;
      int row = chunk >> 2, off = (chunk & 3) * 8;
      *(v8h*)&Ash[0][row * 32 + off] =
          *(const v8h*)(A + (size_t)(mb + row) * K + off);
    }
  }
#endif

  int cur = 0;
  for (int k0 = 0; k0 < K; k0 += 32) {
#if HAS_TDM
    if (wave == 0) __builtin_amdgcn_s_wait_tensorcnt(0);
#endif
    __syncthreads();                              // slab `cur` ready for all
    const int nxt = cur ^ 1;
    if (k0 + 32 < K) {                            // prefetch next slab (async)
#if HAS_TDM
      if (wave == 0)
        tdm_issue_a(A + (size_t)mb * K + k0 + 32, K, M, &Ash[nxt][0]);
#else
      int t = threadIdx.x;
#pragma unroll
      for (int c = 0; c < 2; ++c) {
        int chunk = t * 2 + c;
        int row = chunk >> 2, off = (chunk & 3) * 8;
        *(v8h*)&Ash[nxt][row * 32 + off] =
            *(const v8h*)(A + (size_t)(mb + row) * K + k0 + 32 + off);
      }
#endif
    }

    v16h a0 = load_a_frag(&Ash[cur][0], 32, mloc, 0);      // ds_load_b128 x2
    v16h a1 = load_a_frag(&Ash[cur][0], 32, mloc + 16, 0);
    v16h b[4];
#pragma unroll
    for (int j = 0; j < 4; ++j) b[j] = load_b_frag(Bt, K, n0 + j * 16, k0);
    if (k0 + 32 < K)                               // global_prefetch_b8
      __builtin_prefetch(Bt + (size_t)(n0 + lane) * K + k0 + 32, 0, 1);
#pragma unroll
    for (int j = 0; j < 4; ++j) {
      acc[0][j] = wmma_f16(a0, b[j], acc[0][j]);
      acc[1][j] = wmma_f16(a1, b[j], acc[1][j]);
    }
    cur = nxt;
  }
  // C/D layout: VGPR slot s -> row = s + 8*(lane>>4); col = lane&15
  const int rbase = mb + mloc + (lane >> 4) * 8;
  const int col = n0 + (lane & 15);
#pragma unroll
  for (int i = 0; i < 2; ++i)
#pragma unroll
    for (int j = 0; j < 4; ++j)
#pragma unroll
      for (int s = 0; s < 8; ++s) {
        int row = rbase + i * 16 + s;
        C[(size_t)row * N + col + j * 16] = (OUT)acc[i][j][s];
      }
}

// ---------- RoPE applied in-place to Q and K (fp16) ----------
__global__ __launch_bounds__(256) void rope_kernel(half_t* __restrict__ Q,
                                                   half_t* __restrict__ Kk) {
  int t = blockIdx.x * 256 + threadIdx.x;          // one (even,odd) pair
  int row = t >> 9;
  int pc = t & 511;
  int col = pc * 2;
  int i = (col & 63) >> 1;
  int s = row & (SEQ - 1);
  float inv = __expf(-(float)(2 * i) * (9.210340371976184f / 64.0f));
  float ang = (float)s * inv;
  float sn, cs;
  __sincosf(ang, &sn, &cs);
  size_t o = (size_t)row * D_MODEL + col;
  float q1 = (float)Q[o], q2 = (float)Q[o + 1];
  Q[o]     = (half_t)(q1 * cs - q2 * sn);
  Q[o + 1] = (half_t)(q1 * sn + q2 * cs);
  float k1 = (float)Kk[o], k2 = (float)Kk[o + 1];
  Kk[o]     = (half_t)(k1 * cs - k2 * sn);
  Kk[o + 1] = (half_t)(k1 * sn + k2 * cs);
}

// ---------- flash attention: one wave per 16-row query tile ----------
__global__ __launch_bounds__(32) void flash_attn(const half_t* __restrict__ Q,
                                                 const half_t* __restrict__ K,
                                                 const half_t* __restrict__ Vt,
                                                 half_t* __restrict__ O) {
  __shared__ __align__(16) half_t Pls[16 * 32];
  const int lane = threadIdx.x & 31;
  const int l16 = lane & 15, hi = lane >> 4;
  const int q0 = blockIdx.x * 16;
  const int b = blockIdx.y >> 4;
  const int h = blockIdx.y & 15;
  const half_t* Qb  = Q  + (size_t)b * SEQ * D_MODEL + h * HD;
  const half_t* Kb  = K  + (size_t)b * SEQ * D_MODEL + h * HD;
  const half_t* Vtb = Vt + (size_t)(b * NH + h) * HD * SEQ;   // [d][s]

  v16h qf0 = load_a_frag(Qb, D_MODEL, q0, 0);
  v16h qf1 = load_a_frag(Qb, D_MODEL, q0, 32);
  float mrow[8], lrow[8];
#pragma unroll
  for (int s = 0; s < 8; ++s) { mrow[s] = -3.0e38f; lrow[s] = 0.0f; }
  v8f o[4] = {};
  const float scale = 0.03125f;                    // 1/sqrt(1024)

  for (int kt = 0; kt < q0 + 16; kt += 32) {       // causal key bound
    // ---- S = Q * K^T ----
    v8f sc[2] = {};
#pragma unroll
    for (int c = 0; c < 2; ++c) {
      v16h kf0 = load_b_frag(Kb, D_MODEL, kt + c * 16, 0);
      v16h kf1 = load_b_frag(Kb, D_MODEL, kt + c * 16, 32);
      sc[c] = wmma_f16(qf0, kf0, sc[c]);
      sc[c] = wmma_f16(qf1, kf1, sc[c]);
    }
    // ---- scale (+ causal mask only on diagonal tiles) ----
    if (kt + 31 <= q0) {
      sc[0] = sc[0] * scale;
      sc[1] = sc[1] * scale;
    } else {
#pragma unroll
      for (int s = 0; s < 8; ++s) {
        int qq = q0 + s + hi * 8;
#pragma unroll
        for (int c = 0; c < 2; ++c) {
          float v = (float)sc[c][s] * scale;
          int kk = kt + c * 16 + l16;
          sc[c][s] = (kk > qq) ? -3.0e38f : v;
        }
      }
    }
    // ---- row max / online softmax (16-lane groups share a row) ----
    float rmax[8];
#pragma unroll
    for (int s = 0; s < 8; ++s) {
      float r = fmaxf((float)sc[0][s], (float)sc[1][s]);
#pragma unroll
      for (int off = 1; off < 16; off <<= 1) r = fmaxf(r, __shfl_xor(r, off, 16));
      rmax[s] = r;
    }
#pragma unroll
    for (int s = 0; s < 8; ++s) {
      float mnew = fmaxf(mrow[s], rmax[s]);
      float alpha = __expf(mrow[s] - mnew);
      mrow[s] = mnew;
      float p0 = __expf((float)sc[0][s] - mnew);
      float p1 = __expf((float)sc[1][s] - mnew);
      sc[0][s] = p0; sc[1][s] = p1;
      float rs = p0 + p1;
#pragma unroll
      for (int off = 1; off < 16; off <<= 1) rs += __shfl_xor(rs, off, 16);
      lrow[s] = lrow[s] * alpha + rs;
#pragma unroll
      for (int j = 0; j < 4; ++j) o[j][s] *= alpha;
    }
    // ---- P: C-layout regs -> LDS -> A-layout fragment ----
#pragma unroll
    for (int s = 0; s < 8; ++s) {
      int m = s + hi * 8;
      Pls[m * 32 + l16]      = (half_t)(float)sc[0][s];
      Pls[m * 32 + 16 + l16] = (half_t)(float)sc[1][s];
    }
    __syncthreads();
    v16h pf;
    {
      const half_t* pp = &Pls[l16 * 32 + hi * 8];
      v8h x = *(const v8h*)pp;
      v8h y = *(const v8h*)(pp + 16);
#pragma unroll
      for (int i = 0; i < 8; ++i) { pf[i] = x[i]; pf[i + 8] = y[i]; }
    }
    __syncthreads();
    // ---- O += P * V : Vt is [d][s] so B fragments are two b128 loads ----
#pragma unroll
    for (int j = 0; j < 4; ++j)
      o[j] = wmma_f16(pf, load_b_frag(Vtb, SEQ, j * 16, kt), o[j]);
  }
  // ---- normalize and store merged [b, q, h*64+d] fp16 ----
#pragma unroll
  for (int s = 0; s < 8; ++s) {
    int row = q0 + s + hi * 8;
    float invl = 1.0f / lrow[s];
#pragma unroll
    for (int j = 0; j < 4; ++j)
      O[((size_t)b * SEQ + row) * D_MODEL + h * HD + j * 16 + l16] =
          (half_t)(o[j][s] * invl);
  }
}

// ---------- host launcher ----------
extern "C" void kernel_launch(void* const* d_in, const int* in_sizes, int n_in,
                              void* d_out, int out_size, void* d_ws, size_t ws_size,
                              hipStream_t stream) {
  (void)in_sizes; (void)n_in; (void)out_size; (void)ws_size;
  const half_t* x  = (const half_t*)d_in[0];
  const half_t* qw = (const half_t*)d_in[1];
  const half_t* kw = (const half_t*)d_in[2];
  const half_t* vw = (const half_t*)d_in[3];
  const half_t* ow = (const half_t*)d_in[4];
  float* out = (float*)d_out;

  const size_t MN = (size_t)BATCH * SEQ * D_MODEL;   // 4M fp16
  const size_t WW = (size_t)D_MODEL * D_MODEL;       // 1M fp16
  half_t* ws  = (half_t*)d_ws;
  half_t* Qb  = ws;
  half_t* Kb  = Qb + MN;
  half_t* Vb  = Kb + MN;
  half_t* Mg  = Vb + MN;
  half_t* Vtb = Mg + MN;
  half_t* WqT = Vtb + MN;
  half_t* WkT = WqT + WW;
  half_t* WvT = WkT + WW;
  half_t* WoT = WvT + WW;

  transpose_w<<<4096, 256, 0, stream>>>(qw, WqT);
  transpose_w<<<4096, 256, 0, stream>>>(kw, WkT);
  transpose_w<<<4096, 256, 0, stream>>>(vw, WvT);
  transpose_w<<<4096, 256, 0, stream>>>(ow, WoT);

  dim3 g(32, 8);  // 4096/128 x 1024/128
  gemm_wmma<half_t><<<g, 256, 0, stream>>>(x, WqT, Qb, BATCH * SEQ, D_MODEL, D_MODEL);
  gemm_wmma<half_t><<<g, 256, 0, stream>>>(x, WkT, Kb, BATCH * SEQ, D_MODEL, D_MODEL);
  gemm_wmma<half_t><<<g, 256, 0, stream>>>(x, WvT, Vb, BATCH * SEQ, D_MODEL, D_MODEL);

  v_transpose<<<(int)(MN / 256), 256, 0, stream>>>(Vb, Vtb);
  rope_kernel<<<(BATCH * SEQ * (D_MODEL / 2)) / 256, 256, 0, stream>>>(Qb, Kb);

  dim3 fg(SEQ / 16, BATCH * NH);
  flash_attn<<<fg, 32, 0, stream>>>(Qb, Kb, Vtb, Mg);

  gemm_wmma<float><<<g, 256, 0, stream>>>(Mg, WoT, out, BATCH * SEQ, D_MODEL, D_MODEL);
}
